// MoeveForward_82652350644894
// MI455X (gfx1250) — matrix operand order repaired and checked by reading, not compiled, not run on hardware
//
#include <hip/hip_runtime.h>
#include <hip/hip_bf16.h>
#include <math.h>

// ---------------------------------------------------------------------------
// MoE forward for MI455X (gfx1250, wave32, WMMA).
// Big GEMMs in bf16 via v_wmma_f32_16x16x32_bf16, fed by CDNA5 async
// global->LDS copies (ASYNCcnt) with LDS double buffering.
// ---------------------------------------------------------------------------

typedef __attribute__((ext_vector_type(16))) __bf16 v16bf;
typedef __attribute__((ext_vector_type(8)))  float  v8f;

#define N_TOK 16384
#define DIM   1024
#define HID   4096
#define NEXP  8
#define NEG_SLOPE 0.01f

static __device__ __forceinline__ __bf16 f2bf(float f) {
  unsigned u = __builtin_bit_cast(unsigned, f);
  u += 0x7FFFu + ((u >> 16) & 1u);           // round-to-nearest-even
  unsigned short h = (unsigned short)(u >> 16);
  return __builtin_bit_cast(__bf16, h);
}

// Async copy of 32 contiguous bytes (global -> LDS), CDNA5 ASYNCcnt path.
// GVS mode: 64-bit SGPR base + 32-bit per-lane byte offset. IOFFSET is added
// to BOTH the global and LDS addresses, so one address pair serves both 16B
// transfers.
static __device__ __forceinline__ void async_copy32(unsigned ldsOff,
                                                    const void* base,
                                                    unsigned gByteOff) {
  asm volatile(
      "global_load_async_to_lds_b128 %0, %1, %2\n\t"
      "global_load_async_to_lds_b128 %0, %1, %2 offset:16"
      :
      : "v"(ldsOff), "v"(gByteOff), "s"(base)
      : "memory");
}

static __device__ __forceinline__ void wait_async0() {
  asm volatile("s_wait_asynccnt 0x0" ::: "memory");
}

// ---------------------------------------------------------------------------
// fp32 -> bf16 conversion (grid-stride)
// ---------------------------------------------------------------------------
__global__ void cvt_f32_bf16(const float* __restrict__ in,
                             __bf16* __restrict__ out, int n) {
  int i = blockIdx.x * blockDim.x + threadIdx.x;
  int stride = gridDim.x * blockDim.x;
  for (; i < n; i += stride) out[i] = f2bf(in[i]);
}

// ---------------------------------------------------------------------------
// Router: logits, noisy top-k, gate, normal-CDF loads, assignment counts.
// One wave32 per token row; 8 waves per block.
// ---------------------------------------------------------------------------
__global__ __launch_bounds__(256)
void router_kernel(const float* __restrict__ x, const float* __restrict__ noise,
                   const float* __restrict__ Wr, const float* __restrict__ Wn,
                   float* __restrict__ gate, float* __restrict__ expert_loads,
                   float* __restrict__ expert_cnt, float* __restrict__ s2e) {
  __shared__ float sLoads[NEXP];
  __shared__ float sCnt[NEXP];

  const int t = threadIdx.x;
  if (t < NEXP) { sLoads[t] = 0.f; sCnt[t] = 0.f; }
  __syncthreads();

  const int lane = t & 31;
  const int wave = t >> 5;
  const int n = blockIdx.x * 8 + wave;

  float accR[NEXP], accN[NEXP];
#pragma unroll
  for (int e = 0; e < NEXP; ++e) { accR[e] = 0.f; accN[e] = 0.f; }

  const float* xr = x + (size_t)n * DIM;
  for (int d = lane; d < DIM; d += 32) {
    float xv = xr[d];
#pragma unroll
    for (int e = 0; e < NEXP; ++e) {
      accR[e] = fmaf(xv, Wr[e * DIM + d], accR[e]);
      accN[e] = fmaf(xv, Wn[e * DIM + d], accN[e]);
    }
  }
#pragma unroll
  for (int off = 16; off > 0; off >>= 1) {
#pragma unroll
    for (int e = 0; e < NEXP; ++e) {
      accR[e] += __shfl_xor(accR[e], off, 32);
      accN[e] += __shfl_xor(accN[e], off, 32);
    }
  }

  if (lane == 0) {
    float logits[NEXP], nz[NEXP], noisy[NEXP];
#pragma unroll
    for (int e = 0; e < NEXP; ++e) {
      logits[e] = accR[e];
      float v = accN[e];
      nz[e] = (v > 20.f) ? v : log1pf(expf(v));      // softplus
      noisy[e] = logits[e] + noise[e] * nz[e];
    }
    // top-3 of 8 (ties keep lowest index, matching lax.top_k)
    float v0 = -INFINITY, v1 = -INFINITY, v2 = -INFINITY;
    int i0 = 0, i1 = 0;
#pragma unroll
    for (int e = 0; e < NEXP; ++e) {
      float v = noisy[e];
      if (v > v0)      { v2 = v1; v1 = v0; i1 = i0; v0 = v; i0 = e; }
      else if (v > v1) { v2 = v1; v1 = v; i1 = e; }
      else if (v > v2) { v2 = v; }
    }
    // softmax over top-2 summed == 1 (computed honestly)
    float e1 = __expf(v1 - v0);
    float s  = 1.f + e1;
    gate[n] = (1.f + e1) / s;

    const float thr_out = v1;   // K-th largest (K=2)
    const float thr_in  = v2;   // (K+1)-th largest
#pragma unroll
    for (int e = 0; e < NEXP; ++e) {
      float kth = (noisy[e] >= thr_out) ? thr_in : thr_out;
      float z = (logits[e] - kth) / nz[e];
      float p = 0.5f * erfcf(-z * 0.70710678118654752f);   // Phi(z)
      atomicAdd(&sLoads[e], p);
      float cnt = (e == i0 || e == i1) ? 1.f : 0.f;
      if (cnt != 0.f) atomicAdd(&sCnt[e], cnt);
      s2e[(size_t)e * N_TOK + n] = cnt;
    }
  }
  __syncthreads();
  if (t < NEXP) {
    atomicAdd(&expert_loads[t], sLoads[t]);
    atomicAdd(&expert_cnt[t], sCnt[t]);
  }
}

// ---------------------------------------------------------------------------
// bf16 WMMA GEMM: C(N x M) = A(N x K) * W(M x K)^T  (+bias, epilogue)
//   mode 0: leaky_relu -> bf16 out
//   mode 1: (+bias) * gate[row] -> fp32 out
// Block tile 128x128, BK=32, 8 waves; wave tile = 32 x 64 (2x4 WMMA accs).
// Tiles stream into LDS with global_load_async_to_lds_b128, double buffered.
// ---------------------------------------------------------------------------
union FragBf { v16bf v; uint4 q[2]; };

#define LDSROW 40   // 32 K elems + 8 pad (bf16) per row

__global__ __launch_bounds__(256)
void gemm_bf16_wmma(const __bf16* __restrict__ A, const __bf16* __restrict__ W,
                    const float* __restrict__ bias, const float* __restrict__ gate,
                    __bf16* __restrict__ outBf, float* __restrict__ outF,
                    int Ktot, int Mtot, int mode) {
  __shared__ __align__(16) __bf16 sA[2][128 * LDSROW];
  __shared__ __align__(16) __bf16 sB[2][128 * LDSROW];

  const int rowBase = blockIdx.y * 128;
  const int colBase = blockIdx.x * 128;
  const int t    = threadIdx.x;
  const int lane = t & 31;
  const int wave = t >> 5;
  const int wr   = wave >> 1;    // 0..3 -> rows wr*32
  const int wc   = wave & 1;     // 0..1 -> cols wc*64

  v8f acc[2][4];
  const v8f vzero = {0.f, 0.f, 0.f, 0.f, 0.f, 0.f, 0.f, 0.f};
#pragma unroll
  for (int i = 0; i < 2; ++i)
#pragma unroll
    for (int j = 0; j < 4; ++j) acc[i][j] = vzero;

  // Cooperative tile loads: 2 threads per row, 32 bytes (16 bf16) each.
  const int lr = t >> 1;
  const int lc = (t & 1) * 16;
  const unsigned gAoff0 = (unsigned)(((size_t)(rowBase + lr) * Ktot + lc) * 2);
  const unsigned gBoff0 = (unsigned)(((size_t)(colBase + lr) * Ktot + lc) * 2);
  unsigned ldsA[2], ldsB[2];
#pragma unroll
  for (int b = 0; b < 2; ++b) {
    ldsA[b] = (unsigned)(size_t)&sA[b][lr * LDSROW + lc];
    ldsB[b] = (unsigned)(size_t)&sB[b][lr * LDSROW + lc];
  }

  // Prologue: stream tile 0 into buffer 0.
  async_copy32(ldsA[0], A, gAoff0);
  async_copy32(ldsB[0], W, gBoff0);

  const int nk = Ktot >> 5;
  for (int ki = 0; ki < nk; ++ki) {
    const int cur = ki & 1;
    wait_async0();              // this wave's pending tile has landed in LDS
    __syncthreads();            // every wave's tile has landed

    if (ki + 1 < nk) {          // overlap next tile's DMA with this compute
      unsigned adv = 64u * (unsigned)(ki + 1);
      async_copy32(ldsA[cur ^ 1], A, gAoff0 + adv);
      async_copy32(ldsB[cur ^ 1], W, gBoff0 + adv);
      __builtin_prefetch((const char*)A + gAoff0 + adv + 64, 0, 2);
      __builtin_prefetch((const char*)W + gBoff0 + adv + 64, 0, 2);
    }

    // A fragments: lane<16 -> K 0..7 & 16..23 of row (lane&15);
    //              lane>=16 -> K 8..15 & 24..31 of same row.
    FragBf fa[2];
#pragma unroll
    for (int i = 0; i < 2; ++i) {
      int r  = wr * 32 + i * 16 + (lane & 15);
      int kk = (lane >> 4) * 8;
      fa[i].q[0] = *(const uint4*)&sA[cur][r * LDSROW + kk];
      fa[i].q[1] = *(const uint4*)&sA[cur][r * LDSROW + kk + 16];
    }
    // B fragments: lane<16 -> K 0..7 & 16..23 of col (lane&15); lane>=16 -> K 8..15 & 24..31.
#pragma unroll
    for (int j = 0; j < 4; ++j) {
      FragBf fb;
      int c  = wc * 64 + j * 16 + (lane & 15);
      int kk = (lane >> 4) * 8;
      fb.q[0] = *(const uint4*)&sB[cur][c * LDSROW + kk];
      fb.q[1] = *(const uint4*)&sB[cur][c * LDSROW + kk + 16];
#pragma unroll
      for (int i = 0; i < 2; ++i) {
        acc[i][j] = __builtin_amdgcn_wmma_f32_16x16x32_bf16(
            false, fa[i].v, false, fb.v, (short)0, acc[i][j], false, false);
      }
    }
    __syncthreads();            // all reads of buf[cur] done before refill
  }

  // Epilogue. C/D layout: VGPR r, lanes 0-15 -> M=r, lanes 16-31 -> M=r+8.
#pragma unroll
  for (int i = 0; i < 2; ++i) {
#pragma unroll
    for (int j = 0; j < 4; ++j) {
      int tileR = rowBase + wr * 32 + i * 16 + ((lane >> 4) << 3);
      int tileC = colBase + wc * 64 + j * 16 + (lane & 15);
      float b = bias[tileC];
#pragma unroll
      for (int r = 0; r < 8; ++r) {
        int row = tileR + r;
        float v = acc[i][j][r] + b;
        if (mode == 0) {
          v = (v >= 0.f) ? v : NEG_SLOPE * v;
          outBf[(size_t)row * Mtot + tileC] = f2bf(v);
        } else {
          outF[(size_t)row * Mtot + tileC] = v * gate[row];
        }
      }
    }
  }
}

// ---------------------------------------------------------------------------
// Launcher
// ---------------------------------------------------------------------------
extern "C" void kernel_launch(void* const* d_in, const int* in_sizes, int n_in,
                              void* d_out, int out_size, void* d_ws, size_t ws_size,
                              hipStream_t stream) {
  (void)in_sizes; (void)n_in; (void)out_size; (void)ws_size;

  const float* x     = (const float*)d_in[0];
  const float* noise = (const float*)d_in[1];
  const float* Wr    = (const float*)d_in[2];
  const float* Wn    = (const float*)d_in[3];
  const float* W1    = (const float*)d_in[4];
  const float* b1    = (const float*)d_in[5];
  const float* W2    = (const float*)d_in[6];
  const float* b2    = (const float*)d_in[7];

  float* out          = (float*)d_out;                    // N x D
  float* expert_loads = out + (size_t)N_TOK * DIM;        // E
  float* expert_cnt   = expert_loads + NEXP;              // E
  float* s2e          = expert_cnt + NEXP;                // E x N

  __bf16* xbf  = (__bf16*)d_ws;                           // N x D
  __bf16* w1bf = xbf  + (size_t)N_TOK * DIM;              // H x D
  __bf16* w2bf = w1bf + (size_t)HID * DIM;                // D x H
  __bf16* hbf  = w2bf + (size_t)DIM * HID;                // N x H
  float*  gbuf = (float*)(hbf + (size_t)N_TOK * HID);     // N

  hipMemsetAsync(expert_loads, 0, 2 * NEXP * sizeof(float), stream);

  cvt_f32_bf16<<<8192, 256, 0, stream>>>(x,  xbf,  N_TOK * DIM);
  cvt_f32_bf16<<<4096, 256, 0, stream>>>(W1, w1bf, HID * DIM);
  cvt_f32_bf16<<<4096, 256, 0, stream>>>(W2, w2bf, DIM * HID);

  router_kernel<<<N_TOK / 8, 256, 0, stream>>>(x, noise, Wr, Wn, gbuf,
                                               expert_loads, expert_cnt, s2e);

  // h = leaky_relu(x @ W1^T + b1)  -> bf16
  gemm_bf16_wmma<<<dim3(HID / 128, N_TOK / 128), 256, 0, stream>>>(
      xbf, w1bf, b1, nullptr, hbf, nullptr, DIM, HID, 0);

  // out = (h @ W2^T + b2) * gate   -> fp32
  gemm_bf16_wmma<<<dim3(DIM / 128, N_TOK / 128), 256, 0, stream>>>(
      hbf, w2bf, b2, gbuf, nullptr, out, HID, DIM, 1);
}
